// MambaDecoder_55061480735022
// MI455X (gfx1250) — matrix operand order repaired and checked by reading, not compiled, not run on hardware
//
#include <hip/hip_runtime.h>
#include <stdint.h>

#define D_MODEL 768
#define N_LAYER 8
#define D_INNER 1536
#define D_STATE 16
#define DT_RANK 48
#define D_CONV  4
#define BATCH   4
#define SEQLEN  1024
#define NTOK    (BATCH * SEQLEN)      // 4096 tokens
#define DBC_DIM (DT_RANK + 2 * D_STATE)  // 80
#define RET_TOK 256

typedef __attribute__((ext_vector_type(16))) __bf16 v16bf;
typedef __attribute__((ext_vector_type(8)))  float  v8f;
typedef __attribute__((ext_vector_type(4)))  unsigned int u32x4;
typedef __attribute__((ext_vector_type(8)))  unsigned int u32x8;

__device__ __forceinline__ unsigned short f2bf(float f) {
  unsigned int u = __float_as_uint(f);
  return (unsigned short)((u + 0x7FFFu + ((u >> 16) & 1u)) >> 16);
}
__device__ __forceinline__ float sigmoidf_(float x) { return 1.0f / (1.0f + __expf(-x)); }

// per-lane async global -> LDS 128-bit copy (ASYNCcnt)
__device__ __forceinline__ void async_g2l_b128(unsigned int lds_off, const void* gptr) {
  unsigned long long ga = (unsigned long long)(uintptr_t)gptr;
  asm volatile("global_load_async_to_lds_b128 %0, %1, off"
               :: "v"(lds_off), "v"(ga) : "memory");
}
__device__ __forceinline__ void async_wait(int pending) {
  if (pending) asm volatile("s_wait_asynccnt 0x1" ::: "memory");
  else         asm volatile("s_wait_asynccnt 0x0" ::: "memory");
}

// ---------------------------------------------------------------- utilities
__global__ void f32_to_bf16_kernel(const float* __restrict__ src,
                                   unsigned short* __restrict__ dst, int n) {
  int i = blockIdx.x * blockDim.x + threadIdx.x;
  if (i < n) dst[i] = f2bf(src[i]);
}

__global__ void embed_kernel(const float* __restrict__ x, const float* __restrict__ pos,
                             float* __restrict__ hidden, int n) {
  int i = blockIdx.x * blockDim.x + threadIdx.x;
  if (i < n) hidden[i] = x[i] + pos[i];
}

// ---------------------------------------------------- residual + LayerNorm
__global__ void resid_ln_kernel(const float* __restrict__ hidden,
                                float* __restrict__ residual,
                                const float* __restrict__ w, const float* __restrict__ b,
                                unsigned short* __restrict__ h_bf16, int first) {
  __shared__ float s_sum[256];
  __shared__ float s_sq[256];
  int row = blockIdx.x;                        // 0..NTOK-1
  const float* hrow = hidden + (size_t)row * D_MODEL;
  float* rrow = residual + (size_t)row * D_MODEL;
  int t = threadIdx.x;
  float v[3];
  float s = 0.f, sq = 0.f;
#pragma unroll
  for (int k = 0; k < 3; k++) {
    int i = t + k * 256;
    float r = first ? hrow[i] : (hrow[i] + rrow[i]);
    v[k] = r;
    rrow[i] = r;
    s += r;
    sq += r * r;
  }
  s_sum[t] = s; s_sq[t] = sq;
  __syncthreads();
  for (int off = 128; off > 0; off >>= 1) {
    if (t < off) { s_sum[t] += s_sum[t + off]; s_sq[t] += s_sq[t + off]; }
    __syncthreads();
  }
  float mean = s_sum[0] * (1.f / D_MODEL);
  float var  = s_sq[0] * (1.f / D_MODEL) - mean * mean;
  float rstd = rsqrtf(var + 1e-5f);
#pragma unroll
  for (int k = 0; k < 3; k++) {
    int i = t + k * 256;
    float o = (v[k] - mean) * rstd * w[i] + b[i];
    h_bf16[(size_t)row * D_MODEL + i] = f2bf(o);
  }
}

// ------------------------------------------------------------- WMMA GEMM
// C[M,N] f32 = A[M,K] bf16 x B[N,K]^T bf16
// Double-buffered LDS pipeline:
//   A tile (128x32) staged by the Tensor Data Mover (TENSORcnt),
//   B tile (64x32) staged by per-lane async global->LDS copies (ASYNCcnt);
//   tile k+1 streams while tile k feeds v_wmma_f32_16x16x32_bf16.
#define BM 128
#define BN 64
#define BK 32
#define LDA_S 40   // 64B row + 16B TDM pad -> 80B stride (16B aligned)
#define LDB_S 40

__global__ __launch_bounds__(256) void wmma_gemm_bf16_kernel(
    const unsigned short* __restrict__ A,
    const unsigned short* __restrict__ B,
    float* __restrict__ C, int M, int N, int K) {
  __shared__ __align__(16) unsigned short lsA[2][BM * LDA_S];
  __shared__ __align__(16) unsigned short lsB[2][BN * LDB_S];

  int t = threadIdx.x;
  int lane = t & 31;
  int wave = t >> 5;                 // 0..7
  int wm = (wave & 3) * 32;
  int wn = (wave >> 2) * 32;

  int blockM = blockIdx.y * BM;
  int blockN = blockIdx.x * BN;

  v8f zero = {};
  v8f acc[2][2];
#pragma unroll
  for (int i = 0; i < 2; i++)
#pragma unroll
    for (int j = 0; j < 2; j++) acc[i][j] = zero;

  int lr  = lane & 15;
  int kbA = (lane >> 4) * 8;     // A frag: K runs {kbA..kbA+7, kbA+16..kbA+23}
  int ksB = (lane >> 4) * 8;     // B frag: K runs {ksB..ksB+7, ksB+16..ksB+23}

  // B staging: thread t moves 8 halves (b128) of row brow_l, cols bcol..bcol+7
  int bidx   = t * 8;
  int brow_l = bidx >> 5;
  int bcol   = bidx & 31;
  int brow_g = blockN + brow_l;
  if (brow_g >= N) brow_g = N - 1;           // clamp (stores guarded)
  const unsigned short* Bg = B + (size_t)brow_g * K + bcol;

  unsigned int lsA_base[2], lsB_off[2];
#pragma unroll
  for (int s = 0; s < 2; s++) {
    lsA_base[s] = (unsigned int)(uintptr_t)&lsA[s][0];
    lsB_off[s]  = (unsigned int)(uintptr_t)&lsB[s][brow_l * LDB_S + bcol];
  }

  // Tensor DMA descriptor group1 (K-loop invariant):
  //   data_size=2B, pad 4 DW after every 16 DW (-> 40-half LDS stride),
  //   tensor_dim0=K, tensor_dim1=M, tile 32 x 128, dim0 stride = K elems.
  u32x8 g1;
  g1[0] = (1u << 16) | (1u << 20) | (3u << 22) | (3u << 25);
  g1[1] = ((unsigned int)K & 0xFFFFu) << 16;
  g1[2] = ((unsigned int)K >> 16) | (((unsigned int)M & 0xFFFFu) << 16);
  g1[3] = ((unsigned int)M >> 16) | ((unsigned int)BK << 16);
  g1[4] = (unsigned int)BM;
  g1[5] = (unsigned int)K;
  g1[6] = 0u;
  g1[7] = 0u;

  auto stage = [&](int bsel, int kt) {
    if (wave == 0) {   // TDM: one DMA covers the whole 128x32 A tile
      unsigned long long ga =
          (unsigned long long)(uintptr_t)(A + (size_t)blockM * K + kt);
      u32x4 g0;
      g0[0] = 1u;                                   // count=1, user mode
      g0[1] = lsA_base[bsel];                       // LDS dest address
      g0[2] = (unsigned int)ga;                     // global addr [31:0]
      g0[3] = (unsigned int)(ga >> 32) | (2u << 30);// addr [56:32] | type=2
      asm volatile("tensor_load_to_lds %0, %1" :: "s"(g0), "s"(g1) : "memory");
    }
    async_g2l_b128(lsB_off[bsel], Bg + kt);
  };

  stage(0, 0);                         // prologue: fill buffer 0
  int buf = 0;
  for (int kt = 0; kt < K; kt += BK) {
    int has_next = (kt + BK < K) ? 1 : 0;
    if (has_next) stage(buf ^ 1, kt + BK);      // stream next tile
    // release current buffer: leave the in-flight next transfer pending
    if (wave == 0) {
      if (has_next) __builtin_amdgcn_s_wait_tensorcnt(1);
      else          __builtin_amdgcn_s_wait_tensorcnt(0);
    }
    async_wait(has_next);
    if (has_next) {
      __builtin_prefetch(A + (size_t)(blockM + (t >> 1)) * K + kt + 2 * BK, 0, 3);
    }
    __syncthreads();

    const unsigned short* lA = &lsA[buf][0];
    const unsigned short* lB = &lsB[buf][0];
#pragma unroll
    for (int i = 0; i < 2; i++) {
      int ar = wm + i * 16 + lr;
      union { uint4 q[2]; v16bf v; } af;
      af.q[0] = *reinterpret_cast<const uint4*>(&lA[ar * LDA_S + kbA]);
      af.q[1] = *reinterpret_cast<const uint4*>(&lA[ar * LDA_S + kbA + 16]);
#pragma unroll
      for (int j = 0; j < 2; j++) {
        int brow = wn + j * 16 + lr;
        union { uint4 q[2]; v16bf v; } bfr;
        bfr.q[0] = *reinterpret_cast<const uint4*>(&lB[brow * LDB_S + ksB]);
        bfr.q[1] = *reinterpret_cast<const uint4*>(&lB[brow * LDB_S + ksB + 16]);
        acc[i][j] = __builtin_amdgcn_wmma_f32_16x16x32_bf16(
            false, af.v, false, bfr.v, (short)0, acc[i][j], false, false);
      }
    }
    __syncthreads();   // readers done before this buffer is re-written
    buf ^= 1;
  }

#pragma unroll
  for (int i = 0; i < 2; i++) {
#pragma unroll
    for (int j = 0; j < 2; j++) {
      int rbase = blockM + wm + i * 16 + (lane >> 4) * 8;
      int col   = blockN + wn + j * 16 + (lane & 15);
      if (col < N) {
#pragma unroll
        for (int v = 0; v < 8; v++) {
          C[(size_t)(rbase + v) * N + col] = acc[i][j][v];
        }
      }
    }
  }
}

// ---------------------------------------------- depthwise causal conv + SiLU
__global__ void conv_silu_kernel(const float* __restrict__ xz,
                                 const float* __restrict__ cw, const float* __restrict__ cb,
                                 float* __restrict__ u, unsigned short* __restrict__ u_bf,
                                 int n) {
  int i = blockIdx.x * blockDim.x + threadIdx.x;
  if (i >= n) return;
  int d = i % D_INNER;
  int m = i / D_INNER;          // b*SEQLEN + l
  int l = m % SEQLEN;
  float acc = cb[d];
#pragma unroll
  for (int j = 0; j < D_CONV; j++) {
    int li = l - (D_CONV - 1) + j;
    if (li >= 0)
      acc += xz[(size_t)(m - (D_CONV - 1) + j) * (2 * D_INNER) + d] * cw[d * D_CONV + j];
  }
  float s = acc * sigmoidf_(acc);
  u[i] = s;
  u_bf[i] = f2bf(s);
}

// ------------------------------------------------------- dt proj + softplus
__global__ void dt_softplus_kernel(const float* __restrict__ dbc,
                                   const float* __restrict__ dt_w,
                                   const float* __restrict__ dt_b,
                                   float* __restrict__ dt, int n) {
  int i = blockIdx.x * blockDim.x + threadIdx.x;
  if (i >= n) return;
  int d = i % D_INNER;
  int m = i / D_INNER;
  const float* r = dbc + (size_t)m * DBC_DIM;
  const float* w = dt_w + (size_t)d * DT_RANK;
  float s = dt_b[d];
#pragma unroll
  for (int k = 0; k < DT_RANK; k++) s += r[k] * w[k];
  dt[i] = (s > 20.f) ? s : log1pf(__expf(s));
}

// --------------------------------------- selective scan (+ u*D, * SiLU(z))
__global__ void scan_kernel(const float* __restrict__ dt, const float* __restrict__ u,
                            const float* __restrict__ dbc, const float* __restrict__ xz,
                            const float* __restrict__ A_log, const float* __restrict__ Dv,
                            unsigned short* __restrict__ y_bf) {
  int tidg = blockIdx.x * blockDim.x + threadIdx.x;
  if (tidg >= BATCH * D_INNER) return;
  int b = tidg / D_INNER;
  int d = tidg % D_INNER;
  float A[D_STATE], h[D_STATE];
#pragma unroll
  for (int s = 0; s < D_STATE; s++) {
    A[s] = -__expf(A_log[d * D_STATE + s]);
    h[s] = 0.f;
  }
  float Dval = Dv[d];
  for (int l = 0; l < SEQLEN; l++) {
    size_t m = (size_t)b * SEQLEN + l;
    float dtv = dt[m * D_INNER + d];
    float uv  = u[m * D_INNER + d];
    const float* Bp = dbc + m * DBC_DIM + DT_RANK;
    const float* Cp = Bp + D_STATE;
    float xin = dtv * uv;
    float y = 0.f;
#pragma unroll
    for (int s = 0; s < D_STATE; s++) {
      float dA = __expf(dtv * A[s]);
      h[s] = dA * h[s] + xin * Bp[s];
      y += h[s] * Cp[s];
    }
    y += uv * Dval;
    float z = xz[m * (2 * D_INNER) + D_INNER + d];
    float g = y * (z * sigmoidf_(z));
    y_bf[m * D_INNER + d] = f2bf(g);
  }
}

// ------------------------------------------------ final residual + LN + LN
__global__ void final_kernel(const float* __restrict__ hidden,
                             const float* __restrict__ residual,
                             const float* __restrict__ nfw, const float* __restrict__ nfb,
                             const float* __restrict__ dw, const float* __restrict__ db,
                             float* __restrict__ out) {
  __shared__ float s_sum[256];
  __shared__ float s_sq[256];
  int orow = blockIdx.x;                 // 0..BATCH*RET_TOK-1
  int b  = orow / RET_TOK;
  int lo = orow % RET_TOK;
  size_t src = ((size_t)b * SEQLEN + (SEQLEN - RET_TOK) + lo) * D_MODEL;
  int t = threadIdx.x;
  float v[3];
  float s = 0.f, sq = 0.f;
#pragma unroll
  for (int k = 0; k < 3; k++) {
    int i = t + k * 256;
    float r = hidden[src + i] + residual[src + i];
    v[k] = r; s += r; sq += r * r;
  }
  s_sum[t] = s; s_sq[t] = sq;
  __syncthreads();
  for (int off = 128; off > 0; off >>= 1) {
    if (t < off) { s_sum[t] += s_sum[t + off]; s_sq[t] += s_sq[t + off]; }
    __syncthreads();
  }
  float mean = s_sum[0] * (1.f / D_MODEL);
  float var  = s_sq[0] * (1.f / D_MODEL) - mean * mean;
  float rstd = rsqrtf(var + 1e-5f);
  s = 0.f; sq = 0.f;
#pragma unroll
  for (int k = 0; k < 3; k++) {
    int i = t + k * 256;
    float o = (v[k] - mean) * rstd * nfw[i] + nfb[i];
    v[k] = o; s += o; sq += o * o;
  }
  __syncthreads();
  s_sum[t] = s; s_sq[t] = sq;
  __syncthreads();
  for (int off = 128; off > 0; off >>= 1) {
    if (t < off) { s_sum[t] += s_sum[t + off]; s_sq[t] += s_sq[t + off]; }
    __syncthreads();
  }
  float mean2 = s_sum[0] * (1.f / D_MODEL);
  float var2  = s_sq[0] * (1.f / D_MODEL) - mean2 * mean2;
  float rstd2 = rsqrtf(var2 + 1e-5f);
#pragma unroll
  for (int k = 0; k < 3; k++) {
    int i = t + k * 256;
    out[(size_t)orow * D_MODEL + i] = (v[k] - mean2) * rstd2 * dw[i] + db[i];
  }
}

// =========================================================== host launcher
extern "C" void kernel_launch(void* const* d_in, const int* in_sizes, int n_in,
                              void* d_out, int out_size, void* d_ws, size_t ws_size,
                              hipStream_t stream) {
  (void)in_sizes; (void)n_in; (void)out_size; (void)ws_size;

  const float* x       = (const float*)d_in[0];
  const float* pos     = (const float*)d_in[1];
  const float* norm_w  = (const float*)d_in[2];
  const float* norm_b  = (const float*)d_in[3];
  const float* in_proj = (const float*)d_in[4];
  const float* conv_w  = (const float*)d_in[5];
  const float* conv_b  = (const float*)d_in[6];
  const float* x_proj  = (const float*)d_in[7];
  const float* dt_w    = (const float*)d_in[8];
  const float* dt_b    = (const float*)d_in[9];
  const float* A_log   = (const float*)d_in[10];
  const float* Dv      = (const float*)d_in[11];
  const float* out_proj= (const float*)d_in[12];
  const float* normf_w = (const float*)d_in[13];
  const float* normf_b = (const float*)d_in[14];
  const float* dec_w   = (const float*)d_in[15];
  const float* dec_b   = (const float*)d_in[16];
  float* out = (float*)d_out;

  // workspace carve-up
  char* p = (char*)d_ws;
  auto carve = [&](size_t bytes) -> char* {
    char* r = p;
    p += (bytes + 255) & ~(size_t)255;
    return r;
  };
  unsigned short* w_in_bf  = (unsigned short*)carve((size_t)N_LAYER * 2 * D_INNER * D_MODEL * 2);
  unsigned short* w_xp_bf  = (unsigned short*)carve((size_t)N_LAYER * DBC_DIM * D_INNER * 2);
  unsigned short* w_out_bf = (unsigned short*)carve((size_t)N_LAYER * D_MODEL * D_INNER * 2);
  float* hidden   = (float*)carve((size_t)NTOK * D_MODEL * 4);
  float* residual = (float*)carve((size_t)NTOK * D_MODEL * 4);
  unsigned short* h_bf = (unsigned short*)carve((size_t)NTOK * D_MODEL * 2);
  float* xz  = (float*)carve((size_t)NTOK * 2 * D_INNER * 4);
  float* ub  = (float*)carve((size_t)NTOK * D_INNER * 4);
  unsigned short* u_bf = (unsigned short*)carve((size_t)NTOK * D_INNER * 2);
  float* dbc = (float*)carve((size_t)NTOK * DBC_DIM * 4);
  float* dtv = (float*)carve((size_t)NTOK * D_INNER * 4);
  unsigned short* y_bf = (unsigned short*)carve((size_t)NTOK * D_INNER * 2);

  const int thr = 256;

  {
    int n = N_LAYER * 2 * D_INNER * D_MODEL;
    f32_to_bf16_kernel<<<(n + thr - 1) / thr, thr, 0, stream>>>(in_proj, w_in_bf, n);
  }
  {
    int n = N_LAYER * DBC_DIM * D_INNER;
    f32_to_bf16_kernel<<<(n + thr - 1) / thr, thr, 0, stream>>>(x_proj, w_xp_bf, n);
  }
  {
    int n = N_LAYER * D_MODEL * D_INNER;
    f32_to_bf16_kernel<<<(n + thr - 1) / thr, thr, 0, stream>>>(out_proj, w_out_bf, n);
  }
  {
    int n = NTOK * D_MODEL;
    embed_kernel<<<(n + thr - 1) / thr, thr, 0, stream>>>(x, pos, hidden, n);
  }

  for (int l = 0; l < N_LAYER; l++) {
    resid_ln_kernel<<<NTOK, 256, 0, stream>>>(hidden, residual,
                                              norm_w + (size_t)l * D_MODEL,
                                              norm_b + (size_t)l * D_MODEL,
                                              h_bf, l == 0 ? 1 : 0);
    {  // xz = h @ in_proj^T : M=4096, N=3072, K=768
      dim3 g((2 * D_INNER + BN - 1) / BN, NTOK / BM);
      wmma_gemm_bf16_kernel<<<g, 256, 0, stream>>>(
          h_bf, w_in_bf + (size_t)l * 2 * D_INNER * D_MODEL, xz,
          NTOK, 2 * D_INNER, D_MODEL);
    }
    {
      int n = NTOK * D_INNER;
      conv_silu_kernel<<<(n + thr - 1) / thr, thr, 0, stream>>>(
          xz, conv_w + (size_t)l * D_INNER * D_CONV, conv_b + (size_t)l * D_INNER,
          ub, u_bf, n);
    }
    {  // dbc = u @ x_proj^T : M=4096, N=80, K=1536
      dim3 g((DBC_DIM + BN - 1) / BN, NTOK / BM);
      wmma_gemm_bf16_kernel<<<g, 256, 0, stream>>>(
          u_bf, w_xp_bf + (size_t)l * DBC_DIM * D_INNER, dbc,
          NTOK, DBC_DIM, D_INNER);
    }
    {
      int n = NTOK * D_INNER;
      dt_softplus_kernel<<<(n + thr - 1) / thr, thr, 0, stream>>>(
          dbc, dt_w + (size_t)l * D_INNER * DT_RANK, dt_b + (size_t)l * D_INNER, dtv, n);
    }
    {
      int n = BATCH * D_INNER;
      scan_kernel<<<(n + thr - 1) / thr, thr, 0, stream>>>(
          dtv, ub, dbc, xz, A_log + (size_t)l * D_INNER * D_STATE,
          Dv + (size_t)l * D_INNER, y_bf);
    }
    {  // hidden = y @ out_proj^T : M=4096, N=768, K=1536
      dim3 g((D_MODEL + BN - 1) / BN, NTOK / BM);
      wmma_gemm_bf16_kernel<<<g, 256, 0, stream>>>(
          y_bf, w_out_bf + (size_t)l * D_MODEL * D_INNER, hidden,
          NTOK, D_MODEL, D_INNER);
    }
  }

  final_kernel<<<BATCH * RET_TOK, 256, 0, stream>>>(hidden, residual,
                                                    normf_w, normf_b, dec_w, dec_b, out);
}
